// GNN2Transformer_50749333570255
// MI455X (gfx1250) — compile-verified, matrix operand order checked by reading, not run on hardware
//
#include <hip/hip_runtime.h>
#include <hip/hip_bf16.h>

// ---------------- problem constants (from reference) ----------------
#define NNODES 20000
#define NEDGES 320000
#define ETOT   (NEDGES + NNODES)
#define BATCH  32
#define SQ     31          // L+1
#define DM     22          // d_model
#define DH     11          // head dim
#define SKMEM  625         // NNODES / BATCH
#define DFFN   1028
#define MPAD   20096       // NNODES rounded up to 128-row GEMM blocks

typedef __bf16 bf16_t;
typedef __attribute__((ext_vector_type(16))) __bf16 v16bf;
typedef __attribute__((ext_vector_type(8)))  float  v8f;
typedef __attribute__((ext_vector_type(4)))  int    v4i;

#define CDIV(a, b) (((a) + (b) - 1) / (b))

#if __has_builtin(__builtin_amdgcn_global_load_async_to_lds_b128) && \
    __has_builtin(__builtin_amdgcn_s_wait_asynccnt)
#define USE_ASYNC 1
#else
#define USE_ASYNC 0
#endif

// monotonic float<->uint encoding for atomicMax-based segment max
__device__ __forceinline__ unsigned fenc(float f) {
    unsigned u = __float_as_uint(f);
    return (u & 0x80000000u) ? ~u : (u | 0x80000000u);
}
__device__ __forceinline__ float fdec(unsigned e) {
    unsigned u = (e & 0x80000000u) ? (e ^ 0x80000000u) : ~e;
    return __uint_as_float(u);
}
__device__ __forceinline__ float wave_sum(float v) {
#pragma unroll
    for (int o = 16; o > 0; o >>= 1) v += __shfl_down(v, o, 32);
    return v;
}

// ---------------- generic utility kernels ----------------
__global__ void k_zero(float* __restrict__ p, int n) {
    int i = blockIdx.x * blockDim.x + threadIdx.x;
    if (i < n) p[i] = 0.f;
}

__global__ void k_ea_mean(const float* __restrict__ ea, float* __restrict__ out) {
    __shared__ float red[256];
    int j = blockIdx.x;   // column 0..15
    float acc = 0.f;
    for (int e = threadIdx.x; e < NEDGES; e += 256) acc += ea[(size_t)e * 16 + j];
    red[threadIdx.x] = acc;
    __syncthreads();
    for (int s = 128; s > 0; s >>= 1) {
        if (threadIdx.x < s) red[threadIdx.x] += red[threadIdx.x + s];
        __syncthreads();
    }
    if (threadIdx.x == 0) out[j] = red[0] / (float)NEDGES;
}

// stage activations: f32 [nrows x cols] -> bf16 [MPAD x kpad] (zero pad rows + K)
__global__ void k_stage_a(const float* __restrict__ S, bf16_t* __restrict__ D,
                          int nrows, int cols, int kpad, int total) {
    int i = blockIdx.x * blockDim.x + threadIdx.x;
    if (i >= total) return;
    int r = i / kpad, k = i - r * kpad;
    D[i] = (r < nrows && k < cols) ? (bf16_t)S[(size_t)r * cols + k] : (bf16_t)0.f;
}

// stage weights transposed: W [K x Nc] f32 -> Wt [npad x kpad] bf16, Wt[n][k] = W[k][n]
__global__ void k_stage_w(const float* __restrict__ W, bf16_t* __restrict__ D,
                          int K, int Nc, int kpad, int total) {
    int i = blockIdx.x * blockDim.x + threadIdx.x;
    if (i >= total) return;
    int n = i / kpad, k = i - n * kpad;
    D[i] = (n < Nc && k < K) ? (bf16_t)W[(size_t)k * Nc + n] : (bf16_t)0.f;
}

// ------- WMMA bf16 GEMM with LDS-shared, async double-buffered B tiles -------
// block = 256 threads (8 waves): C-tile 128 x (16*NT). B chunk (16*NT cols x 32 K)
// is staged once per block (async -> LDS), each wave re-reads it from LDS.
// A fragments are register-pipelined one k-chunk ahead; all NT B fragments are
// fetched from LDS before the WMMA chain so ds waits overlap the matrix pipe.
template <int NT>
__global__ __launch_bounds__(256) void k_gemm_bf16(
        const bf16_t* __restrict__ A, const bf16_t* __restrict__ Wt,
        const float* __restrict__ bias, float* __restrict__ C,
        int kpad, int Nc, int Mrows) {
    constexpr int NTILE = 16 * NT;
    constexpr int CHUNK = NTILE * 32;              // bf16 elems per B k-chunk
    __shared__ __attribute__((aligned(64))) bf16_t Bs[2 * CHUNK];

    const int tid  = threadIdx.x;
    const int lane = tid & 31;
    const int wv   = tid >> 5;
    const int lo   = lane & 15;
    const int half = lane >> 4;
    const int m0   = blockIdx.x * 128 + wv * 16;
    const int n0   = blockIdx.y * NTILE;
    const int nseg = NT * 64;                      // 16-byte segments per chunk

    auto stageB = [&](int kc, int buf) {
        const int k0 = kc * 32;
        for (int s = tid; s < nseg; s += 256) {
            int nl = s >> 2, seg = s & 3;
            const bf16_t* src = Wt + (size_t)(n0 + nl) * kpad + k0 + seg * 8;
            bf16_t* dst = Bs + buf * CHUNK + nl * 32 + seg * 8;
#if USE_ASYNC
            __builtin_amdgcn_global_load_async_to_lds_b128(
                (__attribute__((address_space(1))) v4i*)(src),
                (__attribute__((address_space(3))) v4i*)(dst), 0, 0);
#else
            *(uint4*)dst = *(const uint4*)src;
#endif
        }
    };

    const int nk = kpad / 32;
    const bf16_t* aprow = A + (size_t)(m0 + lo) * kpad + 16 * half;

    stageB(0, 0);
    v16bf a_cur = *(const v16bf*)(aprow);
    v16bf a_nxt = a_cur;
#if USE_ASYNC
    __builtin_amdgcn_s_wait_asynccnt(0);
#endif
    __syncthreads();

    v8f acc[NT];
#pragma unroll
    for (int t = 0; t < NT; ++t) acc[t] = (v8f){0.f, 0.f, 0.f, 0.f, 0.f, 0.f, 0.f, 0.f};

    for (int kc = 0; kc < nk; ++kc) {
        const int buf = kc & 1;
        const bool more = (kc + 1 < nk);
        if (more) {
            stageB(kc + 1, buf ^ 1);               // DMA next B chunk into other buffer
            a_nxt = *(const v16bf*)(aprow + (kc + 1) * 32);  // pipeline next A frag
        }
        const bf16_t* bb = Bs + buf * CHUNK + 16 * half;
        v16bf bfrag[NT];
#pragma unroll
        for (int t = 0; t < NT; ++t)
            bfrag[t] = *(const v16bf*)(bb + (t * 16 + lo) * 32);
#pragma unroll
        for (int t = 0; t < NT; ++t)
            acc[t] = __builtin_amdgcn_wmma_f32_16x16x32_bf16(
                false, a_cur, false, bfrag[t], (short)0, acc[t], false, false);
        if (more) {
            a_cur = a_nxt;
#if USE_ASYNC
            __builtin_amdgcn_s_wait_asynccnt(0);
#endif
            __syncthreads();                       // next buffer ready, cur buffer free
        }
    }

#pragma unroll
    for (int t = 0; t < NT; ++t) {
        int col = n0 + t * 16 + lo;
        if (col < Nc) {
            float bv = bias[col];
#pragma unroll
            for (int i = 0; i < 8; ++i) {
                int row = m0 + i + 8 * half;
                if (row < Mrows) C[(size_t)row * Nc + col] = acc[t][i] + bv;
            }
        }
    }
}

// ---------------- GATv2 edge kernels ----------------
// wave per edge: logit[e] = att . leaky_relu(xl[src] + xr[dst] + ea@We); segment max
__global__ void k_edge_logit(const int* __restrict__ eidx, const float* __restrict__ eattr,
                             const float* __restrict__ eam, const float* __restrict__ xl,
                             const float* __restrict__ xr, const float* __restrict__ We,
                             const float* __restrict__ att, float* __restrict__ logit,
                             unsigned* __restrict__ menc, int co) {
    int gid = blockIdx.x * blockDim.x + threadIdx.x;
    int e = gid >> 5, lane = gid & 31;
    if (e >= ETOT) return;
    int s, dn;
    const float* eap;
    if (e < NEDGES) { s = eidx[e]; dn = eidx[NEDGES + e]; eap = eattr + (size_t)e * 16; }
    else            { s = dn = e - NEDGES; eap = eam; }
    float eav[16];
#pragma unroll
    for (int j = 0; j < 16; ++j) eav[j] = eap[j];
    const float* xls = xl + (size_t)s * co;
    const float* xrd = xr + (size_t)dn * co;
    float lg = 0.f;
    for (int c = lane; c < co; c += 32) {
        float w = 0.f;
#pragma unroll
        for (int j = 0; j < 16; ++j) w += eav[j] * We[j * co + c];
        float v = xls[c] + xrd[c] + w;
        v = (v > 0.f) ? v : 0.2f * v;
        lg += att[c] * v;
    }
    lg = wave_sum(lg);
    if (lane == 0) {
        logit[e] = lg;
        atomicMax(menc + dn, fenc(lg));
    }
}

__global__ void k_edge_exp(const int* __restrict__ eidx, const float* __restrict__ logit,
                           const unsigned* __restrict__ menc, float* __restrict__ z,
                           float* __restrict__ ev) {
    int e = blockIdx.x * blockDim.x + threadIdx.x;
    if (e >= ETOT) return;
    int dn = (e < NEDGES) ? eidx[NEDGES + e] : (e - NEDGES);
    float x = expf(logit[e] - fdec(menc[dn]));
    ev[e] = x;
    unsafeAtomicAdd(z + dn, x);
}

// wave per edge: out[dst] += (e/z[dst]) * xl[src]
__global__ void k_edge_aggr(const int* __restrict__ eidx, const float* __restrict__ ev,
                            const float* __restrict__ z, const float* __restrict__ xl,
                            float* __restrict__ out, int co) {
    int gid = blockIdx.x * blockDim.x + threadIdx.x;
    int e = gid >> 5, lane = gid & 31;
    if (e >= ETOT) return;
    int s, dn;
    if (e < NEDGES) { s = eidx[e]; dn = eidx[NEDGES + e]; }
    else            { s = dn = e - NEDGES; }
    float alpha = ev[e] / z[dn];
    const float* xls = xl + (size_t)s * co;
    float* od = out + (size_t)dn * co;
    for (int c = lane; c < co; c += 32) unsafeAtomicAdd(od + c, alpha * xls[c]);
}

__global__ void k_bias_relu(float* __restrict__ S, const float* __restrict__ b,
                            int co, int relu, int total) {
    int i = blockIdx.x * blockDim.x + threadIdx.x;
    if (i >= total) return;
    float v = S[i] + b[i % co];
    if (relu) v = fmaxf(v, 0.f);
    S[i] = v;
}

// ---------------- decoder kernels ----------------
// memory[(s*32+b)*22+d] = state[(b*625+s)*22+d]
__global__ void k_reorder_mem(const float* __restrict__ S, float* __restrict__ M) {
    int i = blockIdx.x * blockDim.x + threadIdx.x;
    if (i >= NNODES * DM) return;
    int r = i / DM, d = i - r * DM;
    int s = r / BATCH, b = r - s * BATCH;
    M[i] = S[((size_t)b * SKMEM + s) * DM + d];
}

__global__ void k_token_init(const int* __restrict__ tgt, float* __restrict__ H) {
    int r = blockIdx.x * blockDim.x + threadIdx.x;
    if (r >= SQ * BATCH) return;
    int s = r / BATCH, b = r - s * BATCH;
    int tok = (s == 0) ? 20 : tgt[(s - 1) * BATCH + b];
    const float c = -9.210340371976184f / (float)DM;   // -ln(10000)/D
    for (int d = 0; d < DM; ++d) {
        float div = expf((float)((d >> 1) * 2) * c);
        float ang = (float)s * div;
        float pe = (d & 1) ? cosf(ang) : sinf(ang);
        H[(size_t)r * DM + d] = pe + ((d == tok) ? 1.f : 0.f);
    }
}

__global__ void k_proj(const float* __restrict__ X, const float* __restrict__ W,
                       const float* __restrict__ b, float* __restrict__ Y,
                       int din, int dout, int relu, int total) {
    int i = blockIdx.x * blockDim.x + threadIdx.x;
    if (i >= total) return;
    int r = i / dout, o = i - r * dout;
    float s = b[o];
    const float* xr = X + (size_t)r * din;
    for (int j = 0; j < din; ++j) s += xr[j] * W[(size_t)j * dout + o];
    if (relu) s = fmaxf(s, 0.f);
    Y[i] = s;
}

// thread per (q, b, head); online softmax over Sk keys; layout [S][B][D]
__global__ void k_attn(const float* __restrict__ Q, const float* __restrict__ K,
                       const float* __restrict__ V, float* __restrict__ O,
                       int Sq, int Sk) {
    int i = blockIdx.x * blockDim.x + threadIdx.x;
    if (i >= Sq * BATCH * 2) return;
    int h = i & 1, b = (i >> 1) & 31, q = i >> 6;
    const float* qp = Q + ((size_t)(q * BATCH + b) * DM + h * DH);
    float qv[DH];
#pragma unroll
    for (int d = 0; d < DH; ++d) qv[d] = qp[d];
    float m = -1e30f, lsum = 0.f, acc[DH];
#pragma unroll
    for (int d = 0; d < DH; ++d) acc[d] = 0.f;
    const float scale = 0.30151134457776363f;   // 1/sqrt(11)
    for (int k = 0; k < Sk; ++k) {
        size_t base = (size_t)(k * BATCH + b) * DM + h * DH;
        const float* kp = K + base;
        float s = 0.f;
#pragma unroll
        for (int d = 0; d < DH; ++d) s += qv[d] * kp[d];
        s *= scale;
        float nm = fmaxf(m, s);
        float c0 = expf(m - nm);
        float p = expf(s - nm);
        const float* vp = V + base;
        lsum = lsum * c0 + p;
#pragma unroll
        for (int d = 0; d < DH; ++d) acc[d] = acc[d] * c0 + p * vp[d];
        m = nm;
    }
    float inv = 1.f / lsum;
    float* op = O + ((size_t)(q * BATCH + b) * DM + h * DH);
#pragma unroll
    for (int d = 0; d < DH; ++d) op[d] = acc[d] * inv;
}

// out = LayerNorm(x + y) * g + b, per row of 22
__global__ void k_add_ln(const float* __restrict__ X, const float* __restrict__ Y,
                         const float* __restrict__ g, const float* __restrict__ b,
                         float* __restrict__ O, int rows) {
    int r = blockIdx.x * blockDim.x + threadIdx.x;
    if (r >= rows) return;
    float t[DM], mu = 0.f;
#pragma unroll
    for (int d = 0; d < DM; ++d) { t[d] = X[(size_t)r * DM + d] + Y[(size_t)r * DM + d]; mu += t[d]; }
    mu /= (float)DM;
    float var = 0.f;
#pragma unroll
    for (int d = 0; d < DM; ++d) { float dd = t[d] - mu; var += dd * dd; }
    var /= (float)DM;
    float inv = rsqrtf(var + 1e-5f);
#pragma unroll
    for (int d = 0; d < DM; ++d) O[(size_t)r * DM + d] = (t[d] - mu) * inv * g[d] + b[d];
}

// ---------------- host orchestration ----------------
extern "C" void kernel_launch(void* const* d_in, const int* in_sizes, int n_in,
                              void* d_out, int out_size, void* d_ws, size_t ws_size,
                              hipStream_t stream) {
    (void)in_sizes; (void)n_in; (void)out_size; (void)ws_size;
    const float* x     = (const float*)d_in[0];
    const int*   eidx  = (const int*)d_in[1];
    const float* eattr = (const float*)d_in[2];
    const int*   tgt   = (const int*)d_in[3];

    // workspace carve-out (all 256B aligned)
    size_t off = 0;
    char* base = (char*)d_ws;
    auto alloc = [&](size_t bytes) -> char* {
        char* p = base + off;
        off += (bytes + 255) & ~(size_t)255;
        return p;
    };
    float*    stateB = (float*)alloc((size_t)NNODES * 512 * 4);
    float*    xlb    = (float*)alloc((size_t)NNODES * 512 * 4);
    float*    xrb    = (float*)alloc((size_t)NNODES * 512 * 4);
    bf16_t*   abf    = (bf16_t*)alloc((size_t)MPAD * 512 * 2);
    bf16_t*   wtb    = (bf16_t*)alloc((size_t)512 * 512 * 2);
    float*    logit  = (float*)alloc((size_t)ETOT * 4);
    float*    evb    = (float*)alloc((size_t)ETOT * 4);
    unsigned* menc   = (unsigned*)alloc((size_t)NNODES * 4);
    float*    zb     = (float*)alloc((size_t)NNODES * 4);
    float*    eam    = (float*)alloc(64);
    float*    mem    = (float*)alloc((size_t)NNODES * DM * 4);
    float*    kcb    = (float*)alloc((size_t)NNODES * DM * 4);
    float*    vcb    = (float*)alloc((size_t)NNODES * DM * 4);
    float*    qb     = (float*)alloc((size_t)SQ * BATCH * DM * 4);
    float*    ab     = (float*)alloc((size_t)SQ * BATCH * DM * 4);
    float*    pb     = (float*)alloc((size_t)SQ * BATCH * DM * 4);
    float*    hA     = (float*)alloc((size_t)SQ * BATCH * DM * 4);
    float*    hB     = (float*)alloc((size_t)SQ * BATCH * DM * 4);
    float*    f1     = (float*)alloc((size_t)SQ * BATCH * DFFN * 4);

    // mean edge attr (fill value for self loops)
    k_ea_mean<<<16, 256, 0, stream>>>(eattr, eam);

    static const int CH[9]    = {16, 128, 256, 512, 512, 512, 256, 128, 22};
    static const int RELUF[8] = {1, 1, 0, 1, 1, 1, 1, 0};

    const float* in = x;
    for (int l = 0; l < 8; ++l) {
        const int ci = CH[l], co = CH[l + 1];
        const int kpad = (ci < 32) ? 32 : ci;
        const float* Wl  = (const float*)d_in[4 + l * 7 + 0];
        const float* bl  = (const float*)d_in[4 + l * 7 + 1];
        const float* Wr  = (const float*)d_in[4 + l * 7 + 2];
        const float* br  = (const float*)d_in[4 + l * 7 + 3];
        const float* We  = (const float*)d_in[4 + l * 7 + 4];
        const float* att = (const float*)d_in[4 + l * 7 + 5];
        const float* bv  = (const float*)d_in[4 + l * 7 + 6];

        // stage activations -> bf16 (frees `in` buffer for output accumulation)
        int nA = MPAD * kpad;
        k_stage_a<<<CDIV(nA, 256), 256, 0, stream>>>(in, abf, NNODES, ci, kpad, nA);

        const int NT = (co % 128 == 0) ? 8 : 2;    // 128/256/512 -> 8-wide, 22 -> 2-wide
        const int ntile = 16 * NT;
        const int copad = CDIV(co, ntile) * ntile;
        const int gy = copad / ntile;
        const int nW = copad * kpad;
        dim3 gg(CDIV(MPAD, 128), gy);

        // xl = state @ Wl + bl
        k_stage_w<<<CDIV(nW, 256), 256, 0, stream>>>(Wl, wtb, ci, co, kpad, nW);
        if (NT == 8) k_gemm_bf16<8><<<gg, 256, 0, stream>>>(abf, wtb, bl, xlb, kpad, co, NNODES);
        else         k_gemm_bf16<2><<<gg, 256, 0, stream>>>(abf, wtb, bl, xlb, kpad, co, NNODES);
        // xr = state @ Wr + br
        k_stage_w<<<CDIV(nW, 256), 256, 0, stream>>>(Wr, wtb, ci, co, kpad, nW);
        if (NT == 8) k_gemm_bf16<8><<<gg, 256, 0, stream>>>(abf, wtb, br, xrb, kpad, co, NNODES);
        else         k_gemm_bf16<2><<<gg, 256, 0, stream>>>(abf, wtb, br, xrb, kpad, co, NNODES);

        // edge softmax + aggregation (scatter into stateB)
        k_zero<<<CDIV(NNODES * co, 256), 256, 0, stream>>>(stateB, NNODES * co);
        k_zero<<<CDIV(NNODES, 256), 256, 0, stream>>>((float*)menc, NNODES);
        k_zero<<<CDIV(NNODES, 256), 256, 0, stream>>>(zb, NNODES);
        k_edge_logit<<<CDIV(ETOT, 8), 256, 0, stream>>>(eidx, eattr, eam, xlb, xrb, We, att,
                                                        logit, menc, co);
        k_edge_exp<<<CDIV(ETOT, 256), 256, 0, stream>>>(eidx, logit, menc, zb, evb);
        k_edge_aggr<<<CDIV(ETOT, 8), 256, 0, stream>>>(eidx, evb, zb, xlb, stateB, co);
        k_bias_relu<<<CDIV(NNODES * co, 256), 256, 0, stream>>>(stateB, bv, co, RELUF[l],
                                                                NNODES * co);
        in = stateB;
    }

    // ---------------- decoder ----------------
    k_reorder_mem<<<CDIV(NNODES * DM, 256), 256, 0, stream>>>(stateB, mem);
    k_token_init<<<CDIV(SQ * BATCH, 256), 256, 0, stream>>>(tgt, hA);

    const int RQ = SQ * BATCH;             // 992 query rows
    const int tQ = RQ * DM;                // elems of a [S,B,D] tensor
    for (int l = 0; l < 4; ++l) {
        const int p = 60 + 26 * l;
        const float* saWq = (const float*)d_in[p + 0];
        const float* saWk = (const float*)d_in[p + 1];
        const float* saWv = (const float*)d_in[p + 2];
        const float* saWo = (const float*)d_in[p + 3];
        const float* sabq = (const float*)d_in[p + 4];
        const float* sabk = (const float*)d_in[p + 5];
        const float* sabv = (const float*)d_in[p + 6];
        const float* sabo = (const float*)d_in[p + 7];
        const float* caWq = (const float*)d_in[p + 8];
        const float* caWk = (const float*)d_in[p + 9];
        const float* caWv = (const float*)d_in[p + 10];
        const float* caWo = (const float*)d_in[p + 11];
        const float* cabq = (const float*)d_in[p + 12];
        const float* cabk = (const float*)d_in[p + 13];
        const float* cabv = (const float*)d_in[p + 14];
        const float* cabo = (const float*)d_in[p + 15];
        const float* W1   = (const float*)d_in[p + 16];
        const float* b1   = (const float*)d_in[p + 17];
        const float* W2   = (const float*)d_in[p + 18];
        const float* b2   = (const float*)d_in[p + 19];
        const float* l1g  = (const float*)d_in[p + 20];
        const float* l1b  = (const float*)d_in[p + 21];
        const float* l2g  = (const float*)d_in[p + 22];
        const float* l2b  = (const float*)d_in[p + 23];
        const float* l3g  = (const float*)d_in[p + 24];
        const float* l3b  = (const float*)d_in[p + 25];

        // self-attention
        k_proj<<<CDIV(tQ, 256), 256, 0, stream>>>(hA, saWq, sabq, qb,  DM, DM, 0, tQ);
        k_proj<<<CDIV(tQ, 256), 256, 0, stream>>>(hA, saWk, sabk, kcb, DM, DM, 0, tQ);
        k_proj<<<CDIV(tQ, 256), 256, 0, stream>>>(hA, saWv, sabv, vcb, DM, DM, 0, tQ);
        k_attn<<<CDIV(RQ * 2, 256), 256, 0, stream>>>(qb, kcb, vcb, ab, SQ, SQ);
        k_proj<<<CDIV(tQ, 256), 256, 0, stream>>>(ab, saWo, sabo, pb, DM, DM, 0, tQ);
        k_add_ln<<<CDIV(RQ, 256), 256, 0, stream>>>(hA, pb, l1g, l1b, hB, RQ);

        // cross-attention against graph memory (Sk=625 per graph)
        const int tM = NNODES * DM;
        k_proj<<<CDIV(tQ, 256), 256, 0, stream>>>(hB, caWq, cabq, qb, DM, DM, 0, tQ);
        k_proj<<<CDIV(tM, 256), 256, 0, stream>>>(mem, caWk, cabk, kcb, DM, DM, 0, tM);
        k_proj<<<CDIV(tM, 256), 256, 0, stream>>>(mem, caWv, cabv, vcb, DM, DM, 0, tM);
        k_attn<<<CDIV(RQ * 2, 256), 256, 0, stream>>>(qb, kcb, vcb, ab, SQ, SKMEM);
        k_proj<<<CDIV(tQ, 256), 256, 0, stream>>>(ab, caWo, cabo, pb, DM, DM, 0, tQ);
        k_add_ln<<<CDIV(RQ, 256), 256, 0, stream>>>(hB, pb, l2g, l2b, hA, RQ);

        // feed-forward
        const int tF = RQ * DFFN;
        k_proj<<<CDIV(tF, 256), 256, 0, stream>>>(hA, W1, b1, f1, DM, DFFN, 1, tF);
        k_proj<<<CDIV(tQ, 256), 256, 0, stream>>>(f1, W2, b2, pb, DFFN, DM, 0, tQ);
        k_add_ln<<<CDIV(RQ, 256), 256, 0, stream>>>(hA, pb, l3g, l3b, hB, RQ);

        float* tmp = hA; hA = hB; hB = tmp;
    }

    // final projection -> d_out [31, 32, 22]
    const float* outW = (const float*)d_in[164];
    const float* outB = (const float*)d_in[165];
    k_proj<<<CDIV(RQ * DM, 256), 256, 0, stream>>>(hA, outW, outB, (float*)d_out,
                                                   DM, DM, 0, RQ * DM);
}
